// ResidualCrossAttention_32908039422517
// MI455X (gfx1250) — matrix operand order repaired and checked by reading, not compiled, not run on hardware
//
#include <hip/hip_runtime.h>
#include <hip/hip_fp16.h>
#include <hip/hip_bf16.h>

#define B_    4
#define C_    128      // C1 == C2 == 128
#define OUT_  256
#define H_    64
#define W_    64
#define HW_   4096
#define EPS_  1e-5f

typedef __attribute__((ext_vector_type(16))) _Float16 v16h;
typedef __attribute__((ext_vector_type(8)))  _Float16 v8h;
typedef __attribute__((ext_vector_type(8)))  float    v8f;

static __device__ __forceinline__ v8f wmma_f16(v16h a, v16h b, v8f c) {
  // D = A(16x32) * B(32x16) + C, f32 accumulate
  return __builtin_amdgcn_wmma_f32_16x16x32_f16(false, a, false, b, (short)0, c, false, false);
}

// ---- fragment loaders (CDNA5 wave32 WMMA layouts, CDNA5 ISA 7.12.2) ----
// A fragment (M=16 x K=32 f16): lane L holds row r=L%16, kbase=(L/16)*8,
// halfs 0..7 -> K=[kb,kb+8), halfs 8..15 -> K=[kb+16,kb+24). Source row-major [row][k].
static __device__ __forceinline__ v16h load_frag_a(const _Float16* src, int ld,
                                                   int row0, int k0, int lane) {
  const int r  = lane & 15;
  const int kb = (lane >> 4) << 3;
  const _Float16* p = src + (size_t)(row0 + r) * ld + k0 + kb;
  const v8h lo = *(const v8h*)(p);
  const v8h hi = *(const v8h*)(p + 16);
  v16h a;
#pragma unroll
  for (int i = 0; i < 8; ++i) { a[i] = lo[i]; a[8 + i] = hi[i]; }
  return a;
}

// B fragment (K=32 x N=16 f16): lane L holds column n=L%16, kb=(L/16)*8,
// halfs 0..7 -> K=[kb,kb+8), halfs 8..15 -> K=[kb+16,kb+24).
// Source is "transposed" storage T[n][k] (leading dim ld): B[k][n] = T[n0+n][k0+k].
static __device__ __forceinline__ v16h load_frag_b(const _Float16* src, int ld,
                                                   int n0, int k0, int lane) {
  const int n  = lane & 15;
  const int kb = (lane >> 4) << 3;
  const _Float16* p = src + (size_t)(n0 + n) * ld + k0 + kb;
  const v8h lo = *(const v8h*)(p);
  const v8h hi = *(const v8h*)(p + 16);
  v16h b;
#pragma unroll
  for (int i = 0; i < 8; ++i) { b[i] = lo[i]; b[8 + i] = hi[i]; }
  return b;
}

static __device__ __forceinline__ v8h cvt8(v8f a) {
  v8h r;
#pragma unroll
  for (int i = 0; i < 8; ++i) r[i] = (_Float16)a[i];
  return r;
}

// ---------------- kernel 0: f32 -> f16 weight conversion ----------------
__global__ void k_f32_to_f16(const float* __restrict__ src, _Float16* __restrict__ dst, int n) {
  int i = blockIdx.x * 256 + threadIdx.x;
  if (i < n) dst[i] = (_Float16)src[i];
}

// ---------------- kernel 1: BN + depthwise 3x3 (SAME) + f16 transpose ----------------
// Block computes a 32(p) x 128(c) tile, stages it in LDS, writes coalesced b128 rows:
// Xt[b][p][c] = dwconv(bn(F)) and Ft[b][p][c] = F (raw, residual GEMM operand).
__global__ void k_prep(const float* __restrict__ F,
                       const float* __restrict__ g,  const float* __restrict__ bta,
                       const float* __restrict__ mu, const float* __restrict__ var,
                       const float* __restrict__ wdw, const float* __restrict__ bdw,
                       _Float16* __restrict__ Xt, _Float16* __restrict__ Ft) {
  __shared__ __align__(16) _Float16 Xs[32 * C_];
  __shared__ __align__(16) _Float16 Fs[32 * C_];
  const int tid = threadIdx.x;          // 256 threads
  const int bb  = blockIdx.y;
  const int p0  = blockIdx.x * 32;
  const int pl  = tid & 31;             // local position (coalesced global reads)
  const int cg  = tid >> 5;             // 0..7
  const int p   = p0 + pl;
  const int row = p >> 6, col = p & 63;

#pragma unroll 4
  for (int ci = 0; ci < C_ / 8; ++ci) {
    const int c = ci * 8 + cg;
    const float scale = g[c] * rsqrtf(var[c] + EPS_);
    const float shift = bta[c] - mu[c] * scale;
    const float* Fc = F + ((size_t)bb * C_ + c) * HW_;

    float acc = bdw[c];
#pragma unroll
    for (int dy = 0; dy < 3; ++dy) {
      const int yy = row + dy - 1;
#pragma unroll
      for (int dx = 0; dx < 3; ++dx) {
        const int xx = col + dx - 1;
        if (yy >= 0 && yy < H_ && xx >= 0 && xx < W_) {
          const float x = Fc[yy * W_ + xx] * scale + shift;  // BN before zero-pad
          acc += wdw[c * 9 + dy * 3 + dx] * x;
        }
      }
    }
    Xs[pl * C_ + c] = (_Float16)acc;
    Fs[pl * C_ + c] = (_Float16)Fc[p];
  }
  __syncthreads();

  // coalesced writeback: 8 threads per p-row, 16 halfs (32B) each
  const int r  = tid >> 3;
  const int sg = tid & 7;
  const size_t dofs = ((size_t)bb * HW_ + p0 + r) * C_ + sg * 16;
  const _Float16* sx = &Xs[r * C_ + sg * 16];
  const _Float16* sf = &Fs[r * C_ + sg * 16];
  ((v8h*)(Xt + dofs))[0] = ((const v8h*)sx)[0];
  ((v8h*)(Xt + dofs))[1] = ((const v8h*)sx)[1];
  ((v8h*)(Ft + dofs))[0] = ((const v8h*)sf)[0];
  ((v8h*)(Ft + dofs))[1] = ((const v8h*)sf)[1];
}

// ---------------- kernel 2a: 1x1 conv GEMM, transposed output [p][o] (Q/K) ----------
__global__ void k_gemm_qkt(const _Float16* __restrict__ W, const float* __restrict__ bias,
                           const _Float16* __restrict__ Xt, _Float16* __restrict__ Out) {
  const int lane = threadIdx.x;
  const int o0 = blockIdx.y * 16;
  const int bb = blockIdx.z;
  const _Float16* Xb = Xt + (size_t)bb * HW_ * C_;

  v16h aw[4];
#pragma unroll
  for (int kk = 0; kk < 4; ++kk) aw[kk] = load_frag_a(W, C_, o0, kk * 32, lane);

  const int g2 = lane >> 4;
  float bs[8];
#pragma unroll
  for (int v = 0; v < 8; ++v) bs[v] = bias[o0 + v + 8 * g2];

#pragma unroll
  for (int pt = 0; pt < 4; ++pt) {
    const int p0 = blockIdx.x * 64 + pt * 16;
    v8f acc;
#pragma unroll
    for (int v = 0; v < 8; ++v) acc[v] = bs[v];
#pragma unroll
    for (int kk = 0; kk < 4; ++kk)
      acc = wmma_f16(aw[kk], load_frag_b(Xb, C_, p0, kk * 32, lane), acc);

    _Float16* dst = Out + (size_t)bb * HW_ * C_ + (size_t)(p0 + (lane & 15)) * C_ + o0 + 8 * g2;
    *(v8h*)dst = cvt8(acc);
  }
}

// ---------------- kernel 2b: 1x1 conv GEMM, natural output [o][p] (V) ---------------
// LDS-transposes the 16x64 register tile so global stores are contiguous b128.
__global__ void k_gemm_v(const _Float16* __restrict__ W, const float* __restrict__ bias,
                         const _Float16* __restrict__ Xt, _Float16* __restrict__ Out) {
  __shared__ __align__(16) _Float16 Ts[16 * 64];   // [o_local][p_local]
  const int lane = threadIdx.x;
  const int o0 = blockIdx.y * 16;
  const int bb = blockIdx.z;
  const _Float16* Xb = Xt + (size_t)bb * HW_ * C_;

  v16h aw[4];
#pragma unroll
  for (int kk = 0; kk < 4; ++kk) aw[kk] = load_frag_a(W, C_, o0, kk * 32, lane);

  const int g2 = lane >> 4;
  const int nc = lane & 15;
  float bs[8];
#pragma unroll
  for (int v = 0; v < 8; ++v) bs[v] = bias[o0 + v + 8 * g2];

#pragma unroll
  for (int pt = 0; pt < 4; ++pt) {
    const int p0 = blockIdx.x * 64 + pt * 16;
    v8f acc;
#pragma unroll
    for (int v = 0; v < 8; ++v) acc[v] = bs[v];
#pragma unroll
    for (int kk = 0; kk < 4; ++kk)
      acc = wmma_f16(aw[kk], load_frag_b(Xb, C_, p0, kk * 32, lane), acc);
#pragma unroll
    for (int v = 0; v < 8; ++v)
      Ts[(v + 8 * g2) * 64 + pt * 16 + nc] = (_Float16)acc[v];
  }
  __syncthreads();

  const int r  = lane >> 1;        // output row 0..15
  const int hs = lane & 1;         // 32-half segment
  const _Float16* s = &Ts[r * 64 + hs * 32];
  _Float16* d = Out + (size_t)bb * C_ * HW_ + (size_t)(o0 + r) * HW_ + blockIdx.x * 64 + hs * 32;
#pragma unroll
  for (int j = 0; j < 4; ++j) ((v8h*)d)[j] = ((const v8h*)s)[j];
}

// ---------------- kernel 3: per-row softmax stats (flash-style over m) --------------
// Each lane keeps a private running (max,sum) for its column subset m%16==lane%16;
// no shuffles in the hot loop. Cross-lane merge happens once at the end.
// stats layout: [0,2*B*HW) rowmax[att][b][n]; [2*B*HW,...) rcpL[att][b][n].
__global__ void k_rowstats(const _Float16* __restrict__ Qt1, const _Float16* __restrict__ Kt1,
                           const _Float16* __restrict__ Qt2, const _Float16* __restrict__ Kt2,
                           float* __restrict__ stats) {
  const int lane = threadIdx.x;
  const int n0 = blockIdx.x * 16;
  const int bb = blockIdx.y;
  const int att = blockIdx.z;
  const _Float16* Q = (att ? Qt2 : Qt1) + (size_t)bb * HW_ * C_;
  const _Float16* K = (att ? Kt2 : Kt1) + (size_t)bb * HW_ * C_;

  v16h aq[4];
#pragma unroll
  for (int kk = 0; kk < 4; ++kk) aq[kk] = load_frag_a(Q, C_, n0, kk * 32, lane);

  float lmax[8], lsum[8];
#pragma unroll
  for (int v = 0; v < 8; ++v) { lmax[v] = -1e30f; lsum[v] = 0.f; }

  for (int mt = 0; mt < HW_ / 16; ++mt) {
    const int m0 = mt * 16;
    if (mt + 1 < HW_ / 16)
      __builtin_prefetch(K + (size_t)(m0 + 16 + (lane & 15)) * C_, 0, 0);
    v8f s = {};
#pragma unroll
    for (int kk = 0; kk < 4; ++kk)
      s = wmma_f16(aq[kk], load_frag_b(K, C_, m0, kk * 32, lane), s);
#pragma unroll
    for (int v = 0; v < 8; ++v) {
      const float t = s[v];
      const float nm = fmaxf(lmax[v], t);
      lsum[v] = lsum[v] * __expf(lmax[v] - nm) + __expf(t - nm);
      lmax[v] = nm;
    }
  }

  // merge the 16 lane-partials of each half-wave (one merge per row)
#pragma unroll
  for (int v = 0; v < 8; ++v) {
    float m = lmax[v], su = lsum[v];
#pragma unroll
    for (int mask = 1; mask <= 8; mask <<= 1) {
      const float mo = __shfl_xor(m, mask, 32);
      const float so = __shfl_xor(su, mask, 32);
      const float nm = fmaxf(m, mo);
      su = su * __expf(m - nm) + so * __expf(mo - nm);
      m = nm;
    }
    lmax[v] = m; lsum[v] = su;
  }

  if ((lane & 15) == 0) {
    const int g2 = lane >> 4;
    float* rm = stats + ((size_t)att * B_ + bb) * HW_;
    float* rl = stats + (size_t)2 * B_ * HW_ + ((size_t)att * B_ + bb) * HW_;
#pragma unroll
    for (int v = 0; v < 8; ++v) {
      const int n = n0 + v + 8 * g2;
      rm[n] = lmax[v];
      rl[n] = 1.0f / lsum[v];
    }
  }
}

// ---------------- kernel 4: A = V * softmax(S)  (reduce over query rows n) ---------
// One wave produces a 128(c) x 16(m) output tile, streaming n in chunks of 32.
// E[n,m]=exp(S-rowmax[n])*rcpL[n] is built in registers, transposed through LDS into
// the B-fragment layout, and consumed by 8 accumulate-WMMAs against V.
__global__ void k_pass2(const _Float16* __restrict__ Qt1, const _Float16* __restrict__ Kt1,
                        const _Float16* __restrict__ V1,
                        const _Float16* __restrict__ Qt2, const _Float16* __restrict__ Kt2,
                        const _Float16* __restrict__ V2,
                        const float* __restrict__ stats, _Float16* __restrict__ cat_t) {
  const int lane = threadIdx.x;
  const int m0 = blockIdx.x * 16;
  const int bb = blockIdx.y;
  const int att = blockIdx.z;
  const _Float16* Q  = (att ? Qt2 : Qt1) + (size_t)bb * HW_ * C_;
  const _Float16* K  = (att ? Kt2 : Kt1) + (size_t)bb * HW_ * C_;
  const _Float16* Vb = (att ? V2  : V1 ) + (size_t)bb * C_ * HW_;
  const float* rm = stats + ((size_t)att * B_ + bb) * HW_;
  const float* rl = stats + (size_t)2 * B_ * HW_ + ((size_t)att * B_ + bb) * HW_;
  const int cofs = att * C_;   // A_i -> channels [0,128), A_w -> [128,256)

  __shared__ __align__(16) _Float16 Et[16 * 32];   // Et[m][n_local]
  __shared__ float srm[32], srl[32];

  v16h kb[4];
#pragma unroll
  for (int kk = 0; kk < 4; ++kk) kb[kk] = load_frag_b(K, C_, m0, kk * 32, lane);

  v8f acc[8];
#pragma unroll
  for (int ct = 0; ct < 8; ++ct) acc[ct] = (v8f){};

  const int g2 = lane >> 4;
  const int mcol = lane & 15;

  for (int nchunk = 0; nchunk < HW_ / 32; ++nchunk) {
    const int n0 = nchunk * 32;
    if (n0 + 32 < HW_)
      __builtin_prefetch(Q + (size_t)(n0 + 32 + lane) * C_, 0, 0);
    srm[lane] = rm[n0 + lane];
    srl[lane] = rl[n0 + lane];
    __syncthreads();

#pragma unroll
    for (int sub = 0; sub < 2; ++sub) {
      v8f s = {};
#pragma unroll
      for (int kk = 0; kk < 4; ++kk)
        s = wmma_f16(load_frag_a(Q, C_, n0 + 16 * sub, kk * 32, lane), kb[kk], s);
      v8h eh;
#pragma unroll
      for (int v = 0; v < 8; ++v) {
        const int nl = 16 * sub + v + 8 * g2;
        eh[v] = (_Float16)(__expf(s[v] - srm[nl]) * srl[nl]);
      }
      *(v8h*)&Et[mcol * 32 + 16 * sub + 8 * g2] = eh;
    }
    __syncthreads();

    const v16h eb = load_frag_b(&Et[0], 32, 0, 0, lane);
#pragma unroll
    for (int ct = 0; ct < 8; ++ct)
      acc[ct] = wmma_f16(load_frag_a(Vb, HW_, ct * 16, n0, lane), eb, acc[ct]);
    __syncthreads();
  }

  _Float16* dst = cat_t + (size_t)bb * HW_ * OUT_ + (size_t)(m0 + mcol) * OUT_ + cofs + 8 * g2;
#pragma unroll
  for (int ct = 0; ct < 8; ++ct)
    *(v8h*)(dst + ct * 16) = cvt8(acc[ct]);
}

// ---------------- kernel 5: out = Wproj*cat + Wres1*F_i + Wres2*F_w + biases -------
__global__ void k_final(const _Float16* __restrict__ Wp, const _Float16* __restrict__ Wr1,
                        const _Float16* __restrict__ Wr2,
                        const float* __restrict__ bp, const float* __restrict__ br1,
                        const float* __restrict__ br2,
                        const _Float16* __restrict__ cat_t, const _Float16* __restrict__ Fti,
                        const _Float16* __restrict__ Ftw, float* __restrict__ out) {
  const int lane = threadIdx.x;
  const int o0 = blockIdx.y * 16;
  const int bb = blockIdx.z;
  const _Float16* Cb = cat_t + (size_t)bb * HW_ * OUT_;
  const _Float16* Fi = Fti   + (size_t)bb * HW_ * C_;
  const _Float16* Fw = Ftw   + (size_t)bb * HW_ * C_;

  v16h ap[8], ar1[4], ar2[4];
#pragma unroll
  for (int kk = 0; kk < 8; ++kk) ap[kk]  = load_frag_a(Wp,  OUT_, o0, kk * 32, lane);
#pragma unroll
  for (int kk = 0; kk < 4; ++kk) ar1[kk] = load_frag_a(Wr1, C_,   o0, kk * 32, lane);
#pragma unroll
  for (int kk = 0; kk < 4; ++kk) ar2[kk] = load_frag_a(Wr2, C_,   o0, kk * 32, lane);

  const int g2 = lane >> 4;
  float bs[8];
#pragma unroll
  for (int v = 0; v < 8; ++v) {
    const int o = o0 + v + 8 * g2;
    bs[v] = bp[o] + br1[o] + br2[o];
  }

#pragma unroll
  for (int pt = 0; pt < 4; ++pt) {
    const int p0 = blockIdx.x * 64 + pt * 16;
    v8f acc;
#pragma unroll
    for (int v = 0; v < 8; ++v) acc[v] = bs[v];
#pragma unroll
    for (int kk = 0; kk < 8; ++kk)
      acc = wmma_f16(ap[kk],  load_frag_b(Cb, OUT_, p0, kk * 32, lane), acc);
#pragma unroll
    for (int kk = 0; kk < 4; ++kk)
      acc = wmma_f16(ar1[kk], load_frag_b(Fi, C_,   p0, kk * 32, lane), acc);
#pragma unroll
    for (int kk = 0; kk < 4; ++kk)
      acc = wmma_f16(ar2[kk], load_frag_b(Fw, C_,   p0, kk * 32, lane), acc);

    float* dst = out + (size_t)bb * OUT_ * HW_;
#pragma unroll
    for (int v = 0; v < 8; ++v)
      dst[(size_t)(o0 + v + 8 * g2) * HW_ + p0 + (lane & 15)] = acc[v];
  }
}

// ------------------------------------------------------------------------------------
extern "C" void kernel_launch(void* const* d_in, const int* in_sizes, int n_in,
                              void* d_out, int out_size, void* d_ws, size_t ws_size,
                              hipStream_t stream) {
  const float* F_i   = (const float*)d_in[0];
  const float* F_w   = (const float*)d_in[1];
  const float* bn1_g = (const float*)d_in[2];
  const float* bn1_b = (const float*)d_in[3];
  const float* bn1_m = (const float*)d_in[4];
  const float* bn1_v = (const float*)d_in[5];
  const float* bn2_g = (const float*)d_in[6];
  const float* bn2_b = (const float*)d_in[7];
  const float* bn2_m = (const float*)d_in[8];
  const float* bn2_v = (const float*)d_in[9];
  const float* wq1 = (const float*)d_in[10]; const float* bq1 = (const float*)d_in[11];
  const float* wk1 = (const float*)d_in[12]; const float* bk1 = (const float*)d_in[13];
  const float* wv1 = (const float*)d_in[14]; const float* bv1 = (const float*)d_in[15];
  const float* wq2 = (const float*)d_in[16]; const float* bq2 = (const float*)d_in[17];
  const float* wk2 = (const float*)d_in[18]; const float* bk2 = (const float*)d_in[19];
  const float* wv2 = (const float*)d_in[20]; const float* bv2 = (const float*)d_in[21];
  const float* wdw1 = (const float*)d_in[22]; const float* bdw1 = (const float*)d_in[23];
  const float* wdw2 = (const float*)d_in[24]; const float* bdw2 = (const float*)d_in[25];
  const float* wproj = (const float*)d_in[26]; const float* bproj = (const float*)d_in[27];
  const float* wres1 = (const float*)d_in[28]; const float* bres1 = (const float*)d_in[29];
  const float* wres2 = (const float*)d_in[30]; const float* bres2 = (const float*)d_in[31];
  float* out = (float*)d_out;

  // ---- workspace carve-up (~49 MB) ----
  char* ws = (char*)d_ws;
  size_t off = 0;
  auto alloc = [&](size_t bytes) -> void* {
    void* p = ws + off;
    off = (off + bytes + 255) & ~(size_t)255;
    return p;
  };
  const size_t ACT = (size_t)B_ * HW_ * C_ * sizeof(_Float16);   // 4 MB
  _Float16* wq1h = (_Float16*)alloc(C_ * C_ * 2);
  _Float16* wk1h = (_Float16*)alloc(C_ * C_ * 2);
  _Float16* wv1h = (_Float16*)alloc(C_ * C_ * 2);
  _Float16* wq2h = (_Float16*)alloc(C_ * C_ * 2);
  _Float16* wk2h = (_Float16*)alloc(C_ * C_ * 2);
  _Float16* wv2h = (_Float16*)alloc(C_ * C_ * 2);
  _Float16* wph  = (_Float16*)alloc(OUT_ * OUT_ * 2);
  _Float16* wr1h = (_Float16*)alloc(OUT_ * C_ * 2);
  _Float16* wr2h = (_Float16*)alloc(OUT_ * C_ * 2);
  _Float16* Ft_i = (_Float16*)alloc(ACT);
  _Float16* Ft_w = (_Float16*)alloc(ACT);
  _Float16* Xt_i = (_Float16*)alloc(ACT);
  _Float16* Xt_w = (_Float16*)alloc(ACT);
  _Float16* Qt_i = (_Float16*)alloc(ACT);
  _Float16* Kt_i = (_Float16*)alloc(ACT);
  _Float16* Qt_w = (_Float16*)alloc(ACT);
  _Float16* Kt_w = (_Float16*)alloc(ACT);
  _Float16* Vi   = (_Float16*)alloc(ACT);
  _Float16* Vw   = (_Float16*)alloc(ACT);
  _Float16* catT = (_Float16*)alloc((size_t)B_ * HW_ * OUT_ * 2);
  float*    stats = (float*)alloc((size_t)4 * B_ * HW_ * sizeof(float));
  (void)ws_size; (void)in_sizes; (void)n_in; (void)out_size;

  // 0) weight conversion
  const float* wsrc[9] = {wq1, wk1, wv1, wq2, wk2, wv2, wproj, wres1, wres2};
  _Float16*    wdst[9] = {wq1h, wk1h, wv1h, wq2h, wk2h, wv2h, wph, wr1h, wr2h};
  const int    wn[9]   = {C_*C_, C_*C_, C_*C_, C_*C_, C_*C_, C_*C_, OUT_*OUT_, OUT_*C_, OUT_*C_};
  for (int i = 0; i < 9; ++i)
    k_f32_to_f16<<<dim3((wn[i] + 255) / 256), dim3(256), 0, stream>>>(wsrc[i], wdst[i], wn[i]);

  // 1) BN + depthwise + LDS-transposed f16 staging
  dim3 gp(HW_ / 32, B_);
  k_prep<<<gp, 256, 0, stream>>>(F_i, bn1_g, bn1_b, bn1_m, bn1_v, wdw1, bdw1, Xt_i, Ft_i);
  k_prep<<<gp, 256, 0, stream>>>(F_w, bn2_g, bn2_b, bn2_m, bn2_v, wdw2, bdw2, Xt_w, Ft_w);

  // 2) QKV projections (WMMA GEMMs). Q/K transposed [p][c], V natural [c][p].
  dim3 gg(HW_ / 64, C_ / 16, B_);
  k_gemm_qkt<<<gg, 32, 0, stream>>>(wq1h, bq1, Xt_i, Qt_i);
  k_gemm_qkt<<<gg, 32, 0, stream>>>(wk1h, bk1, Xt_i, Kt_i);
  k_gemm_v  <<<gg, 32, 0, stream>>>(wv1h, bv1, Xt_i, Vi);
  k_gemm_qkt<<<gg, 32, 0, stream>>>(wq2h, bq2, Xt_w, Qt_w);
  k_gemm_qkt<<<gg, 32, 0, stream>>>(wk2h, bk2, Xt_w, Kt_w);
  k_gemm_v  <<<gg, 32, 0, stream>>>(wv2h, bv2, Xt_w, Vw);

  // 3) softmax row stats: attn1=(Qi,Kw), attn2=(Qw,Ki)
  dim3 gr(HW_ / 16, B_, 2);
  k_rowstats<<<gr, 32, 0, stream>>>(Qt_i, Kt_w, Qt_w, Kt_i, stats);

  // 4) A_i = Vw*P1, A_w = Vi*P2, written into cat (transposed)
  k_pass2<<<gr, 32, 0, stream>>>(Qt_i, Kt_w, Vw, Qt_w, Kt_i, Vi, stats, catT);

  // 5) final projection + residual 1x1 convs
  dim3 gf(HW_ / 64, OUT_ / 16, B_);
  k_final<<<gf, 32, 0, stream>>>(wph, wr1h, wr2h, bproj, bres1, bres2,
                                 catT, Ft_i, Ft_w, out);
}
